// SpMGAT_13374528160103
// MI455X (gfx1250) — matrix operand order repaired and checked by reading, not compiled, not run on hardware
//
#include <hip/hip_runtime.h>
#include <cstdint>
#include <cstddef>

#define N_NODES 50000
#define N_EDGES 800000
#define F_IN    256
#define HIDF    64
#define N_HEADS 8
#define F1      (N_HEADS * HIDF)   /* 512 */
#define EMB     128
#define ALPHA_L 0.2f
#define EPS_G   1e-16f

typedef __bf16 bf16_t;
typedef __attribute__((ext_vector_type(16))) __bf16 v16bf;
typedef __attribute__((ext_vector_type(8)))  float  v8f;

union FragBF {
    uint4 u4[2];
    v16bf v;
};

// ---------------- prep: fp32 -> bf16 conversions / weight transposes ----------------

__global__ void k_cvt_bf16(const float* __restrict__ in, bf16_t* __restrict__ out, int n) {
    int i = blockIdx.x * blockDim.x + threadIdx.x;
    if (i < n) out[i] = (bf16_t)in[i];
}

// Bt1[c*256 + k] = W1[h][k][f], c = h*64+f  (K-contiguous per output column)
__global__ void k_build_bt1(const float* __restrict__ W1, bf16_t* __restrict__ Bt, int n) {
    int i = blockIdx.x * blockDim.x + threadIdx.x;
    if (i >= n) return;
    int c = i >> 8;        // /256
    int k = i & 255;
    int h = c >> 6, f = c & 63;
    Bt[i] = (bf16_t)W1[((size_t)h * F_IN + k) * HIDF + f];
}

// Bt2[c*512 + k] = Wo[k*128 + c]
__global__ void k_build_bt2(const float* __restrict__ Wo, bf16_t* __restrict__ Bt, int n) {
    int i = blockIdx.x * blockDim.x + threadIdx.x;
    if (i >= n) return;
    int c = i >> 9;
    int k = i & 511;
    Bt[i] = (bf16_t)Wo[(size_t)k * EMB + c];
}

// ---------------- WMMA GEMM: C[M,NC] = A[M,K](bf16) * Bt[NC,K](bf16), f32 accum ----------------
// One wave computes a 16x64 tile: A fragment reused across 4 v_wmma_f32_16x16x32_bf16.
// A frag (16x32 bf16): lanes 0-15 -> M=lane, K=0..15; lanes 16-31 -> M=lane-16, K=16..31.
// B frag (32x16 bf16): lanes 0-15 -> N=lane, K=0..15; lanes 16-31 -> N=lane-16, K=16..31.
__global__ void k_gemm_bf16_wmma(const bf16_t* __restrict__ A,
                                 const bf16_t* __restrict__ Bt,
                                 float* __restrict__ C,
                                 int M, int K, int NC) {
    const int ngroups = NC >> 6;               // 64-wide column groups
    const int gwave   = blockIdx.x * (blockDim.x >> 5) + (threadIdx.x >> 5);
    const int mt      = gwave / ngroups;
    const int ng      = gwave - mt * ngroups;
    if (mt * 16 >= M) return;
    const int lane = threadIdx.x & 31;
    const int half = lane >> 4;
    const int l15  = lane & 15;

    const bf16_t* arow = A  + (size_t)(mt * 16 + l15) * K + half * 16;
    const bf16_t* b0   = Bt + (size_t)(ng * 64 + l15) * K + half * 16;
    const bf16_t* b1   = b0 + (size_t)16 * K;
    const bf16_t* b2   = b0 + (size_t)32 * K;
    const bf16_t* b3   = b0 + (size_t)48 * K;

    v8f acc0 = {0.f, 0.f, 0.f, 0.f, 0.f, 0.f, 0.f, 0.f};
    v8f acc1 = acc0, acc2 = acc0, acc3 = acc0;

    for (int k0 = 0; k0 < K; k0 += 32) {
        FragBF a, f0, f1, f2, f3;
        const uint4* ap = (const uint4*)(arow + k0);
        a.u4[0] = ap[0];              // 8 bf16 (K 16*half+0..7)
        a.u4[1] = ap[1];              // +8 elements (K 16*half+8..15)
        const uint4* p0 = (const uint4*)(b0 + k0); f0.u4[0] = p0[0]; f0.u4[1] = p0[1];
        const uint4* p1 = (const uint4*)(b1 + k0); f1.u4[0] = p1[0]; f1.u4[1] = p1[1];
        const uint4* p2 = (const uint4*)(b2 + k0); f2.u4[0] = p2[0]; f2.u4[1] = p2[1];
        const uint4* p3 = (const uint4*)(b3 + k0); f3.u4[0] = p3[0]; f3.u4[1] = p3[1];
        acc0 = __builtin_amdgcn_wmma_f32_16x16x32_bf16(false, a.v, false, f0.v,
                                                       (short)0, acc0, false, false);
        acc1 = __builtin_amdgcn_wmma_f32_16x16x32_bf16(false, a.v, false, f1.v,
                                                       (short)0, acc1, false, false);
        acc2 = __builtin_amdgcn_wmma_f32_16x16x32_bf16(false, a.v, false, f2.v,
                                                       (short)0, acc2, false, false);
        acc3 = __builtin_amdgcn_wmma_f32_16x16x32_bf16(false, a.v, false, f3.v,
                                                       (short)0, acc3, false, false);
    }
    // C/D layout: VGPR r -> M = r + 8*half, N = l15 (per 16-wide tile)
    float* cbase = C + (size_t)(mt * 16 + half * 8) * NC + ng * 64 + l15;
#pragma unroll
    for (int r = 0; r < 8; ++r) {
        cbase[(size_t)r * NC + 0]  = acc0[r];
        cbase[(size_t)r * NC + 16] = acc1[r];
        cbase[(size_t)r * NC + 32] = acc2[r];
        cbase[(size_t)r * NC + 48] = acc3[r];
    }
}

// ---------------- per-node attention score halves: sl[h,n] = h_row . a_l ----------------

__global__ void k_node_scores(const float* __restrict__ Hmat, const float* __restrict__ avec,
                              float* __restrict__ sl, float* __restrict__ sr,
                              int nNodes, int heads, int hid) {
    int t = blockIdx.x * blockDim.x + threadIdx.x;
    if (t >= nNodes * heads) return;
    int h = t / nNodes;
    int n = t - h * nNodes;
    const float* row = Hmat + (size_t)n * heads * hid + (size_t)h * hid;
    const float* al  = avec + (size_t)h * 2 * hid;
    const float* ar  = al + hid;
    float s0 = 0.f, s1 = 0.f;
    for (int f = 0; f < hid; ++f) { float v = row[f]; s0 += v * al[f]; s1 += v * ar[f]; }
    sl[t] = s0;
    sr[t] = s1;
}

// ---------------- layer-1 edge aggregation: one wave per edge, 8 heads ----------------

__global__ void k_edge_agg1(const int* __restrict__ ei,
                            const float* __restrict__ hall,
                            const float* __restrict__ sl, const float* __restrict__ sr,
                            float* __restrict__ denom, float* __restrict__ num) {
    const int e = blockIdx.x * (blockDim.x >> 5) + (threadIdx.x >> 5);
    if (e >= N_EDGES) return;
    const int lane = threadIdx.x & 31;
    const int s = ei[e];
    const int d = ei[N_EDGES + e];
    const float4* hd = (const float4*)(hall + (size_t)d * F1);
    __builtin_prefetch(hd + lane * 4, 0, 0);        // warm L2->L0 path for the gather
    float ee = 0.f;
    if (lane < N_HEADS) {
        float z  = sl[lane * N_NODES + s] + sr[lane * N_NODES + d];
        float lr = fmaxf(z, ALPHA_L * z);
        ee = __expf(-lr);
        atomicAdd(&denom[lane * N_NODES + s], ee);
    }
    const float eh = __shfl(ee, lane >> 2, 32);     // lane handles features of head lane/4
    float* np = num + (size_t)s * F1 + lane * 16;
#pragma unroll
    for (int j = 0; j < 4; ++j) {
        float4 v = hd[lane * 4 + j];
        atomicAdd(np + 4 * j + 0, eh * v.x);
        atomicAdd(np + 4 * j + 1, eh * v.y);
        atomicAdd(np + 4 * j + 2, eh * v.z);
        atomicAdd(np + 4 * j + 3, eh * v.w);
    }
}

// ---------------- layer-1 finalize: elu(num/denom) -> bf16 x1 ----------------

__global__ void k_finalize1(const float* __restrict__ num, const float* __restrict__ denom,
                            bf16_t* __restrict__ x1b) {
    int i = blockIdx.x * blockDim.x + threadIdx.x;
    if (i >= N_NODES * F1) return;
    int n = i >> 9;
    int c = i & (F1 - 1);
    int h = c >> 6;
    float v = num[i] / (denom[h * N_NODES + n] + EPS_G);
    float e = v > 0.f ? v : expm1f(v);
    x1b[i] = (bf16_t)e;
}

// ---------------- layer-2 edge aggregation: one wave per edge, 128 features ----------------

__global__ void k_edge_agg2(const int* __restrict__ ei, const float* __restrict__ h2,
                            const float* __restrict__ sl, const float* __restrict__ sr,
                            float* __restrict__ denom, float* __restrict__ num,
                            float* __restrict__ ee_out) {
    const int e = blockIdx.x * (blockDim.x >> 5) + (threadIdx.x >> 5);
    if (e >= N_EDGES) return;
    const int lane = threadIdx.x & 31;
    const int s = ei[e];
    const int d = ei[N_EDGES + e];
    const float4* hd = (const float4*)(h2 + (size_t)d * EMB);
    __builtin_prefetch(hd + lane, 0, 0);
    float ee = 0.f;
    if (lane == 0) {
        float z  = sl[s] + sr[d];
        float lr = fmaxf(z, ALPHA_L * z);
        ee = __expf(-lr);
        atomicAdd(&denom[s], ee);
        ee_out[e] = ee;
    }
    const float eh = __shfl(ee, 0, 32);
    float4 v = hd[lane];
    float* np = num + (size_t)s * EMB + lane * 4;
    atomicAdd(np + 0, eh * v.x);
    atomicAdd(np + 1, eh * v.y);
    atomicAdd(np + 2, eh * v.z);
    atomicAdd(np + 3, eh * v.w);
}

__global__ void k_finalize2(const float* __restrict__ num, const float* __restrict__ denom,
                            float* __restrict__ out) {
    int i = blockIdx.x * blockDim.x + threadIdx.x;
    if (i >= N_NODES * EMB) return;
    int n = i >> 7;
    float v = num[i] / (denom[n] + EPS_G);
    out[i] = v > 0.f ? v : expm1f(v);
}

__global__ void k_att(const int* __restrict__ ei, const float* __restrict__ ee,
                      const float* __restrict__ denom, float* __restrict__ att) {
    int e = blockIdx.x * blockDim.x + threadIdx.x;
    if (e >= N_EDGES) return;
    att[e] = ee[e] / (denom[ei[e]] + EPS_G);
}

// ---------------- host launcher ----------------

extern "C" void kernel_launch(void* const* d_in, const int* in_sizes, int n_in,
                              void* d_out, int out_size, void* d_ws, size_t ws_size,
                              hipStream_t stream) {
    const float* x  = (const float*)d_in[0];
    const int*   ei = (const int*)d_in[1];
    const float* W1 = (const float*)d_in[2];
    const float* a1 = (const float*)d_in[3];
    const float* Wo = (const float*)d_in[4];
    const float* ao = (const float*)d_in[5];
    float* out = (float*)d_out;

    char* ws = (char*)d_ws;
    size_t off = 0;
    auto take = [&](size_t bytes) -> char* {
        char* p = ws + off;
        off = (off + bytes + 255) & ~(size_t)255;
        return p;
    };
    bf16_t* xb   = (bf16_t*)take((size_t)N_NODES * F_IN * 2);      // 25.6 MB, reused as h2
    float*  h2   = (float*)xb;                                     // N*128*4 == same size
    bf16_t* Bt1  = (bf16_t*)take((size_t)F1 * F_IN * 2);
    bf16_t* Bt2  = (bf16_t*)take((size_t)EMB * F1 * 2);
    float*  hall = (float*)take((size_t)N_NODES * F1 * 4);         // 102.4 MB, reused as x1b
    bf16_t* x1b  = (bf16_t*)hall;
    float*  sl1  = (float*)take((size_t)N_HEADS * N_NODES * 4);
    float*  sr1  = (float*)take((size_t)N_HEADS * N_NODES * 4);
    float*  dn1  = (float*)take((size_t)N_HEADS * N_NODES * 4);
    float*  num1 = (float*)take((size_t)N_NODES * F1 * 4);         // 102.4 MB, head reused as num2
    float*  num2 = num1;
    float*  sl2  = (float*)take((size_t)N_NODES * 4);
    float*  sr2  = (float*)take((size_t)N_NODES * 4);
    float*  dn2  = (float*)take((size_t)N_NODES * 4);
    float*  ee2  = (float*)take((size_t)N_EDGES * 4);

    // prep
    k_cvt_bf16 <<<(N_NODES * F_IN + 255) / 256, 256, 0, stream>>>(x, xb, N_NODES * F_IN);
    k_build_bt1<<<(F1 * F_IN + 255) / 256,      256, 0, stream>>>(W1, Bt1, F1 * F_IN);
    k_build_bt2<<<(EMB * F1 + 255) / 256,       256, 0, stream>>>(Wo, Bt2, EMB * F1);

    // GEMM1: h_all[N,512] = xb @ Bt1^T   (WMMA bf16, f32 accum; 16x64 per wave)
    {
        int waves  = (N_NODES / 16) * (F1 / 64);   // 3125 * 8
        int blocks = (waves + 7) / 8;
        k_gemm_bf16_wmma<<<blocks, 256, 0, stream>>>(xb, Bt1, hall, N_NODES, F_IN, F1);
    }
    k_node_scores<<<(N_NODES * N_HEADS + 255) / 256, 256, 0, stream>>>(
        hall, a1, sl1, sr1, N_NODES, N_HEADS, HIDF);

    hipMemsetAsync(dn1,  0, (size_t)N_HEADS * N_NODES * 4, stream);
    hipMemsetAsync(num1, 0, (size_t)N_NODES * F1 * 4,      stream);
    k_edge_agg1<<<(N_EDGES + 7) / 8, 256, 0, stream>>>(ei, hall, sl1, sr1, dn1, num1);

    k_finalize1<<<(N_NODES * F1 + 255) / 256, 256, 0, stream>>>(num1, dn1, x1b);

    // GEMM2: h2[N,128] = x1b @ Bt2^T
    {
        int waves  = (N_NODES / 16) * (EMB / 64);  // 3125 * 2
        int blocks = (waves + 7) / 8;
        k_gemm_bf16_wmma<<<blocks, 256, 0, stream>>>(x1b, Bt2, h2, N_NODES, F1, EMB);
    }
    k_node_scores<<<(N_NODES + 255) / 256, 256, 0, stream>>>(h2, ao, sl2, sr2, N_NODES, 1, EMB);

    hipMemsetAsync(dn2,  0, (size_t)N_NODES * 4,       stream);
    hipMemsetAsync(num2, 0, (size_t)N_NODES * EMB * 4, stream);
    k_edge_agg2<<<(N_EDGES + 7) / 8, 256, 0, stream>>>(ei, h2, sl2, sr2, dn2, num2, ee2);

    k_finalize2<<<(N_NODES * EMB + 255) / 256, 256, 0, stream>>>(num2, dn2, out);
    hipMemcpyAsync(out + (size_t)N_NODES * EMB, ei, (size_t)2 * N_EDGES * sizeof(int),
                   hipMemcpyDeviceToDevice, stream);
    k_att<<<(N_EDGES + 255) / 256, 256, 0, stream>>>(
        ei, ee2, dn2, out + (size_t)N_NODES * EMB + 2 * N_EDGES);
}
